// Atten_Head_1417339207762
// MI455X (gfx1250) — compile-verified, matrix operand order checked by reading, not run on hardware
//
#include <hip/hip_runtime.h>
#include <hip/hip_bf16.h>

// ---------------------------------------------------------------------------
// Causal attention head, CDNA5 (gfx1250, wave32) WMMA implementation.
//   B=4, S=4096, D_IN=256, D=64
//   Stage 1: convert W -> f16 transposed          (cvt_w_kernel)
//   Stage 2: Q/K/V projections via WMMA f16       (proj_kernel)
//            Q pre-scaled by 1/sqrt(64); V stored transposed [b][d][s]
//   Stage 3: fused causal flash-attention, WMMA   (attn_kernel)
//            K/V tiles staged via GLOBAL_LOAD_ASYNC_TO_LDS_B128
// ---------------------------------------------------------------------------

typedef __attribute__((ext_vector_type(16))) _Float16 v16h;
typedef __attribute__((ext_vector_type(8)))  _Float16 v8h;
typedef __attribute__((ext_vector_type(8)))  float    v8f;
typedef __attribute__((ext_vector_type(4)))  int      v4i;

#define BATCH 4
#define SLEN  4096
#define DIN   256
#define DHEAD 64
#define NROWS (BATCH * SLEN)           // 16384 flattened rows

#define NEG_BIG (-3.0e38f)

// LDS padded strides (f16 elements); strides are multiples of 16B and give
// conflict-free 16-lane row access patterns.
#define KPAD 72     // K tile rows   [32][72]
#define VPAD 40     // V^T tile rows [64][40]
#define PPAD 40     // P tile rows   [16][40] per wave

// ---------------------------------------------------------------------------
// Async global -> LDS copy (16 bytes per lane), CDNA5 async-DMA path.
// Builtin prototype (from hipcc diagnostic): (v4i AS1*, v4i AS3*, Ii, Ii).
// ---------------------------------------------------------------------------
#if __has_builtin(__builtin_amdgcn_global_load_async_to_lds_b128)
#define HAS_ASYNC_LDS 1
typedef __attribute__((address_space(1))) v4i* gv4i_p;
typedef __attribute__((address_space(3))) v4i* lv4i_p;
__device__ __forceinline__ void async_cp16(const _Float16* g, _Float16* l) {
  __builtin_amdgcn_global_load_async_to_lds_b128((gv4i_p)g, (lv4i_p)l, 0, 0);
}
#else
#define HAS_ASYNC_LDS 0
__device__ __forceinline__ void async_cp16(const _Float16* g, _Float16* l) {
  *(v8h*)l = *(const v8h*)g;   // synchronous fallback
}
#endif

__device__ __forceinline__ void wait_async_lds() {
#if HAS_ASYNC_LDS
#if __has_builtin(__builtin_amdgcn_s_wait_asynccnt)
  __builtin_amdgcn_s_wait_asynccnt(0);
#else
  asm volatile("s_wait_asynccnt 0x0" ::: "memory");
#endif
#endif
}

// Load a 16x32 (or 32x16) f16 WMMA fragment half: two contiguous 8xf16 chunks.
// A/B VGPR packing: elements 0..7 -> K = half*8 + j ; 8..15 -> K = half*8+16+j.
__device__ __forceinline__ v16h frag16(const _Float16* p0, const _Float16* p1) {
  v16h r;
#pragma unroll
  for (int j = 0; j < 8; ++j) { r[j] = p0[j]; r[8 + j] = p1[j]; }
  return r;
}

__device__ __forceinline__ v8f wmma_f16(v16h a, v16h b, v8f c) {
  return __builtin_amdgcn_wmma_f32_16x16x32_f16(false, a, false, b,
                                                (short)0, c, false, false);
}

// ---------------------------------------------------------------------------
// Stage 1: convert the three 256x64 f32 weight matrices into f16, transposed
// to [64][256] so projection B-fragments are contiguous per lane.
// Wt layout: m=0 -> wq, m=1 -> wk, m=2 -> wv.
// ---------------------------------------------------------------------------
__global__ void cvt_w_kernel(const float* __restrict__ wq,
                             const float* __restrict__ wk,
                             const float* __restrict__ wv,
                             _Float16* __restrict__ Wt) {
  int idx = blockIdx.x * blockDim.x + threadIdx.x;  // over 3*64*256
  if (idx >= 3 * DHEAD * DIN) return;
  int m = idx / (DHEAD * DIN);
  int r = idx % (DHEAD * DIN);
  int n = r / DIN;          // output column 0..63
  int k = r % DIN;          // input dim   0..255
  const float* w = (m == 0) ? wq : ((m == 1) ? wk : wv);
  Wt[idx] = (_Float16)w[(long)k * DHEAD + n];
}

// ---------------------------------------------------------------------------
// Stage 2: projections. 128 threads = 4 waves; each wave computes 16 rows of
// Q, K and V (64 cols each). Reduction over DIN=256 = 8 WMMA K-steps.
// Q is pre-scaled by 1/sqrt(64) = 0.125. V is stored transposed: Vt[b][d][s].
// ---------------------------------------------------------------------------
__global__ __launch_bounds__(128) void proj_kernel(
    const float* __restrict__ Xk, const float* __restrict__ Xv,
    const float* __restrict__ Xq, const _Float16* __restrict__ Wt,
    _Float16* __restrict__ Qh, _Float16* __restrict__ Kh,
    _Float16* __restrict__ Vt) {
  const int tid  = threadIdx.x;
  const int wave = tid >> 5;
  const int lane = tid & 31;
  const int half = lane >> 4;
  const int ln   = lane & 15;

  const long r0   = (long)blockIdx.x * 64 + wave * 16;  // first of 16 rows
  const long bb   = r0 >> 12;                           // batch (rows/4096)
  const long sloc = r0 & (SLEN - 1);                    // seq position of row 0

  const float* Xs[3]  = {Xq, Xk, Xv};
  _Float16*    Ys[2]  = {Qh, Kh};
  const float  scl[3] = {0.125f, 1.0f, 1.0f};

#pragma unroll
  for (int t = 0; t < 3; ++t) {
    const float*     X  = Xs[t];
    const _Float16*  wt = Wt + (long)t * DHEAD * DIN;
    v8f acc[4];
#pragma unroll
    for (int c = 0; c < 4; ++c) acc[c] = v8f{};

#pragma unroll
    for (int dc = 0; dc < 8; ++dc) {          // 8 x K=32 over DIN=256
      // A fragment: lane row m = ln, convert f32 -> f16 on the fly.
      const float* xr = X + (r0 + ln) * DIN + dc * 32 + half * 8;
      v16h a;
#pragma unroll
      for (int j = 0; j < 8; ++j) {
        a[j]     = (_Float16)xr[j];
        a[8 + j] = (_Float16)xr[16 + j];
      }
#pragma unroll
      for (int c = 0; c < 4; ++c) {
        const _Float16* wr = wt + (long)(c * 16 + ln) * DIN + dc * 32 + half * 8;
        v16h b = frag16(wr, wr + 16);
        acc[c] = wmma_f16(a, b, acc[c]);
      }
    }

    const float s = scl[t];
    if (t < 2) {
      // Q/K: row-major [row][64]. Element (m = half*8+i, n = ln) per chunk.
#pragma unroll
      for (int c = 0; c < 4; ++c) {
#pragma unroll
        for (int i = 0; i < 8; ++i) {
          Ys[t][(r0 + half * 8 + i) * DHEAD + c * 16 + ln] =
              (_Float16)(acc[c][i] * s);
        }
      }
    } else {
      // V: transposed [b][d][s]; i runs along s -> contiguous 16B store.
#pragma unroll
      for (int c = 0; c < 4; ++c) {
        v8h pack;
#pragma unroll
        for (int i = 0; i < 8; ++i) pack[i] = (_Float16)acc[c][i];
        *(v8h*)&Vt[(bb * DHEAD + c * 16 + ln) * SLEN + sloc + half * 8] = pack;
      }
    }
  }
}

// ---------------------------------------------------------------------------
// Stage 3: fused causal flash attention.
// Block = 128 threads (4 waves); wave w owns q rows [q0, q0+16),
// q0 = blockIdx.x*64 + w*16, batch = blockIdx.y.
// Loop over 32-key blocks; online softmax; O accumulated in f32 fragments.
// K and V^T tiles staged to LDS via async global->LDS DMA.
// ---------------------------------------------------------------------------
__global__ __launch_bounds__(128) void attn_kernel(
    const _Float16* __restrict__ Qh, const _Float16* __restrict__ Kh,
    const _Float16* __restrict__ Vt, float* __restrict__ Out) {
  __shared__ _Float16 sK[32 * KPAD];        // K block, row-major [key][d]
  __shared__ _Float16 sVt[DHEAD * VPAD];    // V block transposed [d][key]
  __shared__ _Float16 sP[4 * 16 * PPAD];    // per-wave P tiles [q][key]

  const int tid  = threadIdx.x;
  const int wave = tid >> 5;
  const int lane = tid & 31;
  const int half = lane >> 4;
  const int ln   = lane & 15;

  const int  q0      = blockIdx.x * 64 + wave * 16;
  const long rowbase = (long)blockIdx.y * SLEN;
  const long vbase   = (long)blockIdx.y * DHEAD * SLEN;

  // Q A-fragments for the wave's 16 rows, 2 chunks of K=32 over d=64.
  v16h qa[2];
#pragma unroll
  for (int dc = 0; dc < 2; ++dc) {
    const _Float16* qr = Qh + (rowbase + q0 + ln) * DHEAD + dc * 32 + half * 8;
    qa[dc] = frag16(qr, qr + 16);
  }

  v8f o[4];
#pragma unroll
  for (int c = 0; c < 4; ++c) o[c] = v8f{};
  float mrow[8], lrow[8];
#pragma unroll
  for (int i = 0; i < 8; ++i) { mrow[i] = NEG_BIG; lrow[i] = 0.0f; }

  _Float16* pw = sP + wave * 16 * PPAD;

  const int nkb = 2 * blockIdx.x + 2;       // key blocks needed by this block

  for (int kbi = 0; kbi < nkb; ++kbi) {
    const int kb = kbi * 32;

    __syncthreads();  // previous tile fully consumed

    // ---- stage K (row-major) and V^T into LDS: 2x 4KB, 8-f16 chunks ----
#pragma unroll
    for (int u = 0; u < 2; ++u) {
      const int c  = tid + u * 128;        // chunk 0..255
      const int kr = c >> 3, ks = c & 7;   // K:  row 0..31, seg 0..7
      const int vd = c >> 2, vs = c & 3;   // V^T: d  0..63, seg 0..3
      const _Float16* kg = Kh + (rowbase + kb + kr) * DHEAD + ks * 8;
      const _Float16* vg = Vt + vbase + (long)vd * SLEN + kb + vs * 8;
      async_cp16(kg, &sK[kr * KPAD + ks * 8]);
      async_cp16(vg, &sVt[vd * VPAD + vs * 8]);
      if (kbi + 1 < nkb) {  // prefetch next block (global_prefetch_b8)
        __builtin_prefetch(kg + 32 * DHEAD, 0, 1);
        __builtin_prefetch(vg + 32, 0, 1);
      }
    }
    wait_async_lds();
    __syncthreads();  // tile visible to all waves

    if (kb <= q0 + 15) {  // wave has unmasked work in this key block
      // ---- S = Q * K^T : 16 q rows x 32 keys ----
      v8f s0 = v8f{}, s1 = v8f{};
#pragma unroll
      for (int dc = 0; dc < 2; ++dc) {
        const _Float16* k0r = &sK[ln * KPAD + dc * 32 + half * 8];
        const _Float16* k1r = &sK[(16 + ln) * KPAD + dc * 32 + half * 8];
        s0 = wmma_f16(qa[dc], frag16(k0r, k0r + 16), s0);
        s1 = wmma_f16(qa[dc], frag16(k1r, k1r + 16), s1);
      }

      // ---- causal mask on the diagonal blocks ----
      if (kb + 31 > q0) {
#pragma unroll
        for (int i = 0; i < 8; ++i) {
          const int m = q0 + half * 8 + i;
          if (kb + ln > m)      s0[i] = NEG_BIG;
          if (kb + 16 + ln > m) s1[i] = NEG_BIG;
        }
      }

      // ---- online softmax (row reductions across 16 lanes per half) ----
#pragma unroll
      for (int i = 0; i < 8; ++i) {
        float mx = fmaxf(s0[i], s1[i]);
        mx = fmaxf(mx, __shfl_xor(mx, 1, 32));
        mx = fmaxf(mx, __shfl_xor(mx, 2, 32));
        mx = fmaxf(mx, __shfl_xor(mx, 4, 32));
        mx = fmaxf(mx, __shfl_xor(mx, 8, 32));
        const float mn    = fmaxf(mrow[i], mx);
        const float alpha = __expf(mrow[i] - mn);
        mrow[i] = mn;
        const float p0 = __expf(s0[i] - mn);
        const float p1 = __expf(s1[i] - mn);
        float rs = p0 + p1;
        rs += __shfl_xor(rs, 1, 32);
        rs += __shfl_xor(rs, 2, 32);
        rs += __shfl_xor(rs, 4, 32);
        rs += __shfl_xor(rs, 8, 32);
        lrow[i] = lrow[i] * alpha + rs;
        o[0][i] *= alpha; o[1][i] *= alpha; o[2][i] *= alpha; o[3][i] *= alpha;
        s0[i] = p0; s1[i] = p1;
      }

      // ---- P (C layout) -> LDS -> reload as A fragment ----
#pragma unroll
      for (int i = 0; i < 8; ++i) {
        pw[(half * 8 + i) * PPAD + ln]      = (_Float16)s0[i];
        pw[(half * 8 + i) * PPAD + 16 + ln] = (_Float16)s1[i];
      }
      asm volatile("s_wait_dscnt 0x0" ::: "memory");  // intra-wave DS ordering
      const _Float16* pr = &pw[ln * PPAD + half * 8];
      v16h pa = frag16(pr, pr + 16);

      // ---- O += P * V ----
#pragma unroll
      for (int c = 0; c < 4; ++c) {
        const _Float16* vr = &sVt[(c * 16 + ln) * VPAD + half * 8];
        o[c] = wmma_f16(pa, frag16(vr, vr + 16), o[c]);
      }
    }
  }

  // ---- normalize and write O (f32) ----
  float invl[8];
#pragma unroll
  for (int i = 0; i < 8; ++i) invl[i] = 1.0f / lrow[i];
#pragma unroll
  for (int c = 0; c < 4; ++c) {
#pragma unroll
    for (int i = 0; i < 8; ++i) {
      Out[(rowbase + q0 + half * 8 + i) * DHEAD + c * 16 + ln] =
          o[c][i] * invl[i];
    }
  }
}

// ---------------------------------------------------------------------------
// Host-side launcher.
// d_in order: inputs_for_keys, inputs_for_values, inputs_for_queries, wq, wk, wv
// ---------------------------------------------------------------------------
extern "C" void kernel_launch(void* const* d_in, const int* in_sizes, int n_in,
                              void* d_out, int out_size, void* d_ws, size_t ws_size,
                              hipStream_t stream) {
  const float* Xk = (const float*)d_in[0];
  const float* Xv = (const float*)d_in[1];
  const float* Xq = (const float*)d_in[2];
  const float* wq = (const float*)d_in[3];
  const float* wk = (const float*)d_in[4];
  const float* wv = (const float*)d_in[5];
  float* Out = (float*)d_out;

  // Workspace layout (f16): Qh | Kh | Vt | Wt
  _Float16* Qh = (_Float16*)d_ws;
  _Float16* Kh = Qh + (size_t)NROWS * DHEAD;
  _Float16* Vt = Kh + (size_t)NROWS * DHEAD;
  _Float16* Wt = Vt + (size_t)NROWS * DHEAD;

  // Stage 1: weight convert/transpose (3 * 64 * 256 elements).
  {
    int total = 3 * DHEAD * DIN;
    cvt_w_kernel<<<(total + 255) / 256, 256, 0, stream>>>(wq, wk, wv, Wt);
  }

  // Stage 2: projections. 256 blocks x 128 threads (4 waves x 16 rows = 64).
  proj_kernel<<<NROWS / 64, 128, 0, stream>>>(Xk, Xv, Xq, Wt, Qh, Kh, Vt);

  // Stage 3: fused causal flash attention. grid (S/64, B).
  attn_kernel<<<dim3(SLEN / 64, BATCH), 128, 0, stream>>>(Qh, Kh, Vt, Out);
}
